// ThreeWayAttention_58703613002463
// MI455X (gfx1250) — matrix operand order repaired and checked by reading, not compiled, not run on hardware
//
#include <hip/hip_runtime.h>

typedef __attribute__((ext_vector_type(16))) _Float16 v16h;
typedef __attribute__((ext_vector_type(8)))  _Float16 v8h;
typedef __attribute__((ext_vector_type(8)))  float    v8f;

#define Bz 32
#define Sz 128
#define Dz 32
#define Fz 32
#define Uz 11
#define NEGm (-1.0e9f)

__global__ __launch_bounds__(256, 1)
void threeway_attn_kernel(const float* __restrict__ X,
                          const float* __restrict__ V,
                          const float* __restrict__ W1, const float* __restrict__ B1,
                          const float* __restrict__ W2, const float* __restrict__ B2,
                          const float* __restrict__ W3, const float* __restrict__ B3,
                          float* __restrict__ out)
{
    // LDS working set (~215 KB of the 320 KB WGP pool)
    __shared__ float     Q[3][Sz][12];        // Q1/Q2/Q3, padded stride 12
    __shared__ _Float16  Ea [Sz][Sz];         // exp(A/9 - Ma)   [i][j]
    __shared__ _Float16  EaT[Sz][Sz];         //                 [j][i]
    __shared__ _Float16  Eb [Sz][Sz];         // exp(Bm/9 - Mb)  [j][k]
    __shared__ _Float16  EbT[Sz][Sz];         //                 [k][j]
    __shared__ _Float16  Ec [Sz][Sz];         // exp(C/9 - Mc)   [i][k]
    __shared__ _Float16  EcT[Sz][Sz];         //                 [k][i]
    __shared__ float     m1un[Sz], m2un[Sz], m3un[Sz];
    __shared__ float     red[3][256];
    __shared__ float     maxv[3];
    __shared__ float     zinv;

    const int b   = blockIdx.x;
    const int tid = threadIdx.x;
    const float* Xb = X + b * Sz * Dz;
    const float* Vb = V + b * Sz * Fz;

    // ---------------- Phase 1: Q_m = X @ W_m + b_m  (K=11, tiny) -------------
    for (int idx = tid; idx < 3 * Sz * Uz; idx += 256) {
        int m   = idx / (Sz * Uz);
        int rem = idx - m * (Sz * Uz);
        int row = rem / Uz;
        int u   = rem - row * Uz;
        const float* w  = (m == 0) ? W1 : ((m == 1) ? W2 : W3);
        const float* bb = (m == 0) ? B1 : ((m == 1) ? B2 : B3);
        const float* xr = Xb + row * Dz;
        float acc = bb[u];
        #pragma unroll
        for (int d = 0; d < Dz; ++d) acc += xr[d] * w[d * Uz + u];
        Q[m][row][u] = acc;
    }
    if (tid < Sz) { m1un[tid] = 0.f; m2un[tid] = 0.f; m3un[tid] = 0.f; }
    __syncthreads();

    // ---------------- Phase 2: per-matrix max of (QpQq^T + diag*NEG)/9 -------
    float lmax0 = -1e30f, lmax1 = -1e30f, lmax2 = -1e30f;
    for (int n = tid; n < 3 * Sz * Sz; n += 256) {
        int m   = n >> 14;
        int rem = n & (Sz * Sz - 1);
        int i   = rem >> 7, j = rem & 127;
        int p = (m == 1) ? 1 : 0;          // A:(Q1,Q2) B:(Q2,Q3) C:(Q1,Q3)
        int q = (m == 0) ? 1 : 2;
        const float* qa = Q[p][i];
        const float* qb = Q[q][j];
        float acc = 0.f;
        #pragma unroll
        for (int u = 0; u < Uz; ++u) acc += qa[u] * qb[u];
        float val = (acc + ((i == j) ? NEGm : 0.f)) * (1.0f / 9.0f);
        if (m == 0)      lmax0 = fmaxf(lmax0, val);
        else if (m == 1) lmax1 = fmaxf(lmax1, val);
        else             lmax2 = fmaxf(lmax2, val);
    }
    red[0][tid] = lmax0; red[1][tid] = lmax1; red[2][tid] = lmax2;
    __syncthreads();
    if (tid < 3) {
        float mx = -1e30f;
        for (int t = 0; t < 256; ++t) mx = fmaxf(mx, red[tid][t]);
        maxv[tid] = mx;
    }
    __syncthreads();

    // ---------------- Phase 3: exponentials -> f16 LDS (both layouts) --------
    for (int n = tid; n < 3 * Sz * Sz; n += 256) {
        int m   = n >> 14;
        int rem = n & (Sz * Sz - 1);
        int i   = rem >> 7, j = rem & 127;
        int p = (m == 1) ? 1 : 0;
        int q = (m == 0) ? 1 : 2;
        const float* qa = Q[p][i];
        const float* qb = Q[q][j];
        float acc = 0.f;
        #pragma unroll
        for (int u = 0; u < Uz; ++u) acc += qa[u] * qb[u];
        float val = (acc + ((i == j) ? NEGm : 0.f)) * (1.0f / 9.0f);
        float e   = __expf(val - maxv[m]);
        _Float16 h = (_Float16)e;
        if (m == 0)      { Ea[i][j] = h; EaT[j][i] = h; }
        else if (m == 1) { Eb[i][j] = h; EbT[j][i] = h; }
        else             { Ec[i][j] = h; EcT[j][i] = h; }
    }
    __syncthreads();

    // ---------------- Phase 4/5: WMMA 128x128x128 contractions ---------------
    const int wave = tid >> 5;
    const int lane = tid & 31;
    const int half = lane >> 4;   // K-half selector
    const int l16  = lane & 15;   // A-row / B-col within tile
    const int ti   = wave;        // each wave owns one 16-row stripe (M)
    const int ibase = ti * 16 + half * 8;

    // Pass G = Ec * Eb^T ; m1_un[i] += Ea[i,j]*G[i,j], m2_un[j] += same
    {
        const _Float16* arow = &Ec[ti * 16 + l16][0];   // invariant across tj
        for (int tj = 0; tj < 8; ++tj) {
            v8f acc = {};
            const _Float16* brow = &Eb[tj * 16 + l16][0];
            #pragma unroll
            for (int kb = 0; kb < 4; ++kb) {
                int k0 = kb * 32;
                v8h alo = *(const v8h*)(arow + k0 + half * 8);
                v8h ahi = *(const v8h*)(arow + k0 + half * 8 + 16);
                v8h blo = *(const v8h*)(brow + k0 + half * 8);
                v8h bhi = *(const v8h*)(brow + k0 + half * 8 + 16);
                v16h av = __builtin_shufflevector(alo, ahi, 0,1,2,3,4,5,6,7,8,9,10,11,12,13,14,15);
                v16h bv = __builtin_shufflevector(blo, bhi, 0,1,2,3,4,5,6,7,8,9,10,11,12,13,14,15);
                acc = __builtin_amdgcn_wmma_f32_16x16x32_f16(false, av, false, bv,
                                                             (short)0, acc, false, false);
            }
            // epilogue: one contiguous b128 load of Ea^T, fire-and-forget ds_add_f32
            v8h ea = *(const v8h*)(&EaT[tj * 16 + l16][ibase]);
            float s2 = 0.f;
            #pragma unroll
            for (int r = 0; r < 8; ++r) {
                float v = (float)ea[r] * acc[r];
                s2 += v;
                atomicAdd(&m1un[ibase + r], v);
            }
            atomicAdd(&m2un[tj * 16 + l16], s2);
        }
    }

    // Pass H = Ea * Eb ; m3_un[k] += Ec[i,k]*H[i,k]
    {
        const _Float16* arow = &Ea[ti * 16 + l16][0];   // invariant across tk
        for (int tk = 0; tk < 8; ++tk) {
            v8f acc = {};
            const _Float16* brow = &EbT[tk * 16 + l16][0];
            #pragma unroll
            for (int kb = 0; kb < 4; ++kb) {
                int k0 = kb * 32;
                v8h alo = *(const v8h*)(arow + k0 + half * 8);
                v8h ahi = *(const v8h*)(arow + k0 + half * 8 + 16);
                v8h blo = *(const v8h*)(brow + k0 + half * 8);
                v8h bhi = *(const v8h*)(brow + k0 + half * 8 + 16);
                v16h av = __builtin_shufflevector(alo, ahi, 0,1,2,3,4,5,6,7,8,9,10,11,12,13,14,15);
                v16h bv = __builtin_shufflevector(blo, bhi, 0,1,2,3,4,5,6,7,8,9,10,11,12,13,14,15);
                acc = __builtin_amdgcn_wmma_f32_16x16x32_f16(false, av, false, bv,
                                                             (short)0, acc, false, false);
            }
            v8h ec = *(const v8h*)(&EcT[tk * 16 + l16][ibase]);
            float s3 = 0.f;
            #pragma unroll
            for (int r = 0; r < 8; ++r) s3 += (float)ec[r] * acc[r];
            atomicAdd(&m3un[tk * 16 + l16], s3);
        }
    }
    __syncthreads();

    // ---------------- Phase 6: Z and outputs ---------------------------------
    if (tid == 0) {
        float z = 0.f;
        for (int s = 0; s < Sz; ++s) z += m1un[s];
        zinv = 1.0f / z;
    }
    __syncthreads();
    if (tid < 96) {
        int which = tid >> 5;          // 0,1,2 -> m1,m2,m3
        int fc    = tid & 31;
        const float* mm = (which == 0) ? m1un : ((which == 1) ? m2un : m3un);
        float acc = 0.f;
        for (int s = 0; s < Sz; ++s) acc += mm[s] * Vb[s * Fz + fc];
        out[b * 96 + tid] = acc * zinv;
    }
}

extern "C" void kernel_launch(void* const* d_in, const int* in_sizes, int n_in,
                              void* d_out, int out_size, void* d_ws, size_t ws_size,
                              hipStream_t stream) {
    const float* X  = (const float*)d_in[0];
    const float* V  = (const float*)d_in[1];
    const float* W1 = (const float*)d_in[2];
    const float* B1 = (const float*)d_in[3];
    const float* W2 = (const float*)d_in[4];
    const float* B2 = (const float*)d_in[5];
    const float* W3 = (const float*)d_in[6];
    const float* B3 = (const float*)d_in[7];
    float* out = (float*)d_out;
    threeway_attn_kernel<<<dim3(Bz), dim3(256), 0, stream>>>(X, V, W1, B1, W2, B2, W3, B3, out);
}
